// Self_Attn_45406394253752
// MI455X (gfx1250) — matrix operand order repaired and checked by reading, not compiled, not run on hardware
//
#include <hip/hip_runtime.h>

// CDNA5 / gfx1250 wave32 WMMA types
typedef __attribute__((ext_vector_type(16))) __bf16 v16bf;
typedef __attribute__((ext_vector_type(8)))  float  v8f;

// Problem constants (from reference): B=16, C=256, W=64, H=64, D=C/8=32
constexpr int NB = 16;
constexpr int NC = 256;
constexpr int NW = 64;
constexpr int NH = 64;
constexpr int ND = 32;

// float -> bf16, round-to-nearest-even (bit trick; always compiles)
__device__ __forceinline__ __bf16 f2bf(float f) {
    unsigned u = __builtin_bit_cast(unsigned, f);
    u += 0x7FFFu + ((u >> 16) & 1u);
    return __builtin_bit_cast(__bf16, (unsigned short)(u >> 16));
}

// Load A fragment (16x32 bf16, M x K) from row-major [M][K] LDS.
// ISA 7.12.2: lanes 0-15 -> M=lane, K = k0+{0..7, 16..23};
//             lanes 16-31 -> M=lane-16, K = k0+{8..15, 24..31}.
__device__ __forceinline__ v16bf load_a(const __bf16* A, int lda, int m0, int k0, int lane) {
    int m  = m0 + (lane & 15);
    int kb = k0 + ((lane & 16) ? 8 : 0);
    const __bf16* p = A + m * lda + kb;
    v16bf a;
#pragma unroll
    for (int e = 0; e < 8; ++e) { a[e] = p[e]; a[e + 8] = p[e + 16]; }
    return a;
}

// Load B fragment (32x16 bf16, K x N) from a TRANSPOSED [N][K] LDS array.
// lanes 0-15 -> N=lane, K = k0+{0..7, 16..23}; lanes 16-31 -> N=lane-16, K = k0+{8..15, 24..31}.
__device__ __forceinline__ v16bf load_bt(const __bf16* Bt, int ldb, int n0, int k0, int lane) {
    int n  = n0 + (lane & 15);
    int kb = k0 + ((lane & 16) ? 8 : 0);
    const __bf16* p = Bt + n * ldb + kb;
    v16bf b;
#pragma unroll
    for (int e = 0; e < 8; ++e) { b[e] = p[e]; b[e + 8] = p[e + 16]; }
    return b;
}

__device__ __forceinline__ v8f wmma_bf16(v16bf a, v16bf b, v8f c) {
    // (neg_a, A, neg_b, B, c_mod, C, reuse_a, reuse_b)
    return __builtin_amdgcn_wmma_f32_16x16x32_bf16(false, a, false, b, (short)0, c,
                                                   false, false);
}

// Software-pipelined GEMM tile: prefetch fragments for k+1 before the WMMA of k,
// so the per-WMMA s_wait_dscnt only covers the *previous* iteration's loads.
template <int K>
__device__ __forceinline__ v8f gemm_tile(const __bf16* A, int lda,
                                         const __bf16* Bt, int ldb,
                                         int m0, int n0, int lane) {
    v8f acc = {0.f, 0.f, 0.f, 0.f, 0.f, 0.f, 0.f, 0.f};
    v16bf a = load_a(A, lda, m0, 0, lane);
    v16bf b = load_bt(Bt, ldb, n0, 0, lane);
#pragma unroll
    for (int k0 = 32; k0 < K + 32; k0 += 32) {
        v16bf an = a, bn = b;
        if (k0 < K) {                       // compile-time folded under unroll
            an = load_a(A, lda, m0, k0, lane);
            bn = load_bt(Bt, ldb, n0, k0, lane);
        }
        acc = wmma_bf16(a, b, acc);
        a = an; b = bn;
    }
    return acc;
}

__global__ __launch_bounds__(256) void self_attn_row_kernel(
    const float* __restrict__ x,   // [B, C, W, H]
    const float* __restrict__ Wq,  // [D, C]
    const float* __restrict__ Wk,  // [D, C]
    float* __restrict__ out)       // [B, C, W, H]
{
    // ---- LDS carve (104 KB). sE/sP alias the weight region (dead after GEMM1/2).
    __shared__ __align__(16) unsigned char smem[106496];
    __bf16* sXt = (__bf16*)(smem);            // [H][C]  A for q/k GEMMs      32 KB
    __bf16* sX  = (__bf16*)(smem + 32768);    // [C][H]  A for out GEMM       32 KB
    __bf16* sQ  = (__bf16*)(smem + 65536);    // [H][D]  q                     4 KB
    __bf16* sKT = (__bf16*)(smem + 69632);    // [H][D]  k^T (= B^T of energy) 4 KB
    __bf16* sWq = (__bf16*)(smem + 73728);    // [D][C]  (= B^T of q GEMM)    16 KB
    __bf16* sWk = (__bf16*)(smem + 90112);    // [D][C]                       16 KB
    float*  sE  = (float*) (smem + 73728);    // [H][H]  energy (aliases sWq) 16 KB
    __bf16* sP  = (__bf16*)(smem + 90112);    // [H][H]  softmax (aliases sWk) 8 KB

    const int tid  = threadIdx.x;
    const int wave = tid >> 5;
    const int lane = tid & 31;
    const int b    = blockIdx.x >> 6;   // / W
    const int w    = blockIdx.x & 63;   // % W

    // ---- Stage x[b,:,w,:] -> bf16, both orientations. One channel per thread.
    const float* px = x + (((size_t)b * NC + tid) * NW + w) * NH;
    {
        const int c = tid;
#pragma unroll
        for (int h = 0; h < NH; h += 4) {
            float4 v = *(const float4*)(px + h);
            __bf16 b0 = f2bf(v.x), b1 = f2bf(v.y), b2 = f2bf(v.z), b3 = f2bf(v.w);
            sX[c * NH + h + 0] = b0;  sX[c * NH + h + 1] = b1;
            sX[c * NH + h + 2] = b2;  sX[c * NH + h + 3] = b3;
            sXt[(h + 0) * NC + c] = b0;  sXt[(h + 1) * NC + c] = b1;
            sXt[(h + 2) * NC + c] = b2;  sXt[(h + 3) * NC + c] = b3;
        }
    }
    // ---- Stage weights (8192 elems each), coalesced.
#pragma unroll
    for (int i = 0; i < 32; ++i) {
        int idx = i * 256 + tid;
        sWq[idx] = f2bf(Wq[idx]);
        sWk[idx] = f2bf(Wk[idx]);
    }
    __syncthreads();

    // ---- GEMM 1/2: q[h][d] and kT[h][d] = (X^T)(W^T). 16 tiles, 2 per wave.
#pragma unroll
    for (int t = 0; t < 2; ++t) {
        int tile = wave * 2 + t;          // 0..15 ; >=8 -> k projection
        bool isK = tile >= 8;
        int tt = tile & 7;
        int m0 = (tt >> 1) * 16;          // h tile (0..48)
        int n0 = (tt & 1) * 16;           // d tile (0,16)
        const __bf16* Bt = isK ? sWk : sWq;
        v8f acc = gemm_tile<NC>(sXt, NC, Bt, NC, m0, n0, lane);
        __bf16* Dst = isK ? sKT : sQ;
        int n  = n0 + (lane & 15);
        int mb = m0 + ((lane & 16) ? 8 : 0);
#pragma unroll
        for (int r = 0; r < 8; ++r) Dst[(mb + r) * ND + n] = f2bf(acc[r]);
    }
    __syncthreads();   // also releases the weight region for sE/sP reuse

    // ---- GEMM 3: energy E[i][j] = q[i][:] . kT[j][:]. 16 tiles, K=32 = 1 WMMA.
#pragma unroll
    for (int t = 0; t < 2; ++t) {
        int tile = wave * 2 + t;          // 0..15 over 4x4 tiles
        int m0 = (tile >> 2) * 16;
        int n0 = (tile & 3) * 16;
        v8f acc = gemm_tile<ND>(sQ, ND, sKT, ND, m0, n0, lane);
        int n  = n0 + (lane & 15);
        int mb = m0 + ((lane & 16) ? 8 : 0);
#pragma unroll
        for (int r = 0; r < 8; ++r) sE[(mb + r) * NH + n] = acc[r];
    }
    __syncthreads();

    // ---- Row softmax over j (unscaled, like the reference). One row per thread.
    if (tid < NH) {
        const float* e = sE + tid * NH;
        float mx = -3.402823466e38f;
        for (int j = 0; j < NH; ++j) mx = fmaxf(mx, e[j]);
        float s = 0.f;
        for (int j = 0; j < NH; ++j) s += __expf(e[j] - mx);
        float inv = 1.0f / s;
        for (int j = 0; j < NH; ++j) sP[tid * NH + j] = f2bf(__expf(e[j] - mx) * inv);
    }
    __syncthreads();

    // Warm the gate-reload row of x (L2-resident; one global_prefetch_b8/thread)
    // so the epilogue loads overlap GEMM4's WMMAs.
    __builtin_prefetch((const void*)px, 0, 3);

    // ---- GEMM 4: out[c][i] = sum_j X[c][j] * P[i][j]; fused gate + store.
    // B^T layout needed is [i][j] == natural P layout.
#pragma unroll
    for (int t = 0; t < 8; ++t) {
        int tile = wave * 8 + t;          // 0..63 over 16x4 tiles
        int m0 = (tile >> 2) * 16;        // c tile
        int n0 = (tile & 3) * 16;         // output-h tile
        v8f acc = gemm_tile<NH>(sX, NH, sP, NH, m0, n0, lane);
        int n  = n0 + (lane & 15);
        int mb = m0 + ((lane & 16) ? 8 : 0);
#pragma unroll
        for (int r = 0; r < 8; ++r) {
            int c = mb + r;
            size_t idx = (((size_t)b * NC + c) * NW + w) * NH + n;
            out[idx] = acc[r] * x[idx];   // elementwise gate; x re-read hits L2
        }
    }
}

extern "C" void kernel_launch(void* const* d_in, const int* in_sizes, int n_in,
                              void* d_out, int out_size, void* d_ws, size_t ws_size,
                              hipStream_t stream) {
    (void)in_sizes; (void)n_in; (void)out_size; (void)d_ws; (void)ws_size;
    const float* x  = (const float*)d_in[0];
    const float* Wq = (const float*)d_in[1];
    const float* Wk = (const float*)d_in[2];
    float* out = (float*)d_out;

    dim3 grid(NB * NW);   // 1024 (b,w) problems, one workgroup each
    dim3 block(256);      // 8 wave32
    self_attn_row_kernel<<<grid, block, 0, stream>>>(x, Wq, Wk, out);
}
